// Attention_15710990369355
// MI455X (gfx1250) — matrix operand order repaired and checked by reading, not compiled, not run on hardware
//
#include <hip/hip_runtime.h>
#include <stdint.h>

typedef _Float16 h16;
typedef __attribute__((ext_vector_type(16))) _Float16 v16h;
typedef __attribute__((ext_vector_type(8)))  _Float16 v8h;
typedef __attribute__((ext_vector_type(8)))  float    v8f;

// ---------------------------------------------------------------------------
// Problem constants
// ---------------------------------------------------------------------------
#define BB   2
#define NN   2048
#define CC   1024
#define HH   16
#define DH   64
#define MROWS (BB*NN)          // 4096
#define N3C  (3*CC)            // 3072

__device__ __forceinline__ v8f zero_v8f() {
  v8f z;
#pragma unroll
  for (int i = 0; i < 8; ++i) z[i] = 0.0f;
  return z;
}

__device__ __forceinline__ v8f wmma_f16(v16h a, v16h b, v8f c) {
  // v_wmma_f32_16x16x32_f16
  return __builtin_amdgcn_wmma_f32_16x16x32_f16(
      /*neg_a=*/false, a, /*neg_b=*/false, b,
      /*c_mod=*/(short)0, c, /*reuse_a=*/false, /*reuse_b=*/false);
}

// A-fragment (16-bit, 16x32): lane group g covers K in [16g,16g+16).
// p points at row base + 16g; loads p[0..7] and p[8..15].
__device__ __forceinline__ v16h load_a16(const h16* p) {
  v8h lo = *(const v8h*)(p);
  v8h hi = *(const v8h*)(p + 8);
  v16h r;
#pragma unroll
  for (int i = 0; i < 8; ++i) { r[i] = lo[i]; r[i + 8] = hi[i]; }
  return r;
}

// ---------------------------------------------------------------------------
// f32 -> f16 flat convert
// ---------------------------------------------------------------------------
__global__ void __launch_bounds__(256) cvt_f32_f16_kernel(const float* __restrict__ src,
                                                          h16* __restrict__ dst, int n) {
  int i = blockIdx.x * 256 + threadIdx.x;
  if (i < n) dst[i] = (h16)src[i];
}

// ---------------------------------------------------------------------------
// Pack f32 weights [K x Ntot] (row major) into WMMA B-fragment tiles:
//   dst[((nt*KT + kt)*32 + lane)*16 + e] = W[kt*32 + 16*(lane>>4) + e][nt*16 + (lane&15)]
// so the GEMM reads one contiguous 32B v16h per lane per fragment.
// ---------------------------------------------------------------------------
__global__ void __launch_bounds__(256) pack_w_kernel(const float* __restrict__ W,
                                                     h16* __restrict__ Wp,
                                                     int K, int Ntot) {
  int t    = blockIdx.x * 256 + threadIdx.x;
  int lane = t & 31;
  int tile = t >> 5;                 // tile = nt*KT + kt
  int KT = K >> 5, NT = Ntot >> 4;
  if (tile >= KT * NT) return;
  int kt = tile % KT;
  int nt = tile / KT;
  int n  = nt * 16 + (lane & 15);
  int k0 = kt * 32 + ((lane >> 4) * 16);
  v16h frag;
#pragma unroll
  for (int e = 0; e < 16; ++e)
    frag[e] = (h16)W[(size_t)(k0 + e) * Ntot + n];
  *(v16h*)(Wp + ((size_t)tile * 32 + lane) * 16) = frag;
}

// ---------------------------------------------------------------------------
// WMMA GEMM:  Cout[M x Ntot] (f32) = A[M x K] (f16 row-major) * Bp (packed) + bias
// Block = 128 threads (4 waves, 2x2), tile 128x128; each wave computes 64x64.
// B panels (8KB per K-step) are streamed into LDS with
// GLOBAL_LOAD_ASYNC_TO_LDS_B128, double-buffered, synced via s_wait_asynccnt +
// split workgroup barriers. A is loaded direct from global (L2-resident),
// with global_prefetch lookahead.
// ---------------------------------------------------------------------------
__global__ void __launch_bounds__(128) gemm_wmma_kernel(const h16* __restrict__ A,
                                                        const h16* __restrict__ Bp,
                                                        const float* __restrict__ bias,
                                                        float* __restrict__ Cout,
                                                        int M, int Ntot, int K) {
  __shared__ h16 sB[2][8 * 512];               // 2 x 8KB double buffer

  const int tid  = threadIdx.x;
  const int lane = tid & 31;
  const int wave = tid >> 5;
  const int wy = wave >> 1, wx = wave & 1;
  const int m0 = blockIdx.y * 128 + wy * 64;
  const int n0 = blockIdx.x * 128;             // block column base (8 n-tiles)
  const int col = lane & 15;
  const int g   = lane >> 4;
  const int kb  = g * 16;
  const int KT  = K >> 5;
  const int nt0 = n0 >> 4;

  // async copy: 8KB panel = 8 tiles x 1KB; per round 2 tiles (128 thr x 16B)
  const int cchunk = tid >> 6;                 // which of the 2 tiles this round
  const int coff   = (tid & 63) * 8;           // halfs within a 1KB tile chunk
  // LDS byte address of sB (low 32 bits of a generic LDS pointer = LDS offset)
  const uint32_t lds_base = (uint32_t)(uintptr_t)(&sB[0][0]);

  auto issue_panel = [&](int kt, int buf) {
#pragma unroll
    for (int r = 0; r < 4; ++r) {
      int tile = 2 * r + cchunk;               // 0..7
      const h16* src = Bp + ((size_t)(nt0 + tile) * KT + kt) * 512 + coff;
      uint32_t dst = lds_base + (uint32_t)((buf * 4096 + tile * 512 + coff) * 2);
      asm volatile("global_load_async_to_lds_b128 %0, %1, off"
                   :: "v"(dst), "v"((uint64_t)(uintptr_t)src)
                   : "memory");
    }
  };

  v8f acc[4][4];
#pragma unroll
  for (int mi = 0; mi < 4; ++mi)
#pragma unroll
    for (int nj = 0; nj < 4; ++nj) acc[mi][nj] = zero_v8f();

  issue_panel(0, 0);                           // prologue: fill buffer 0

  for (int kt = 0; kt < KT; ++kt) {
    if (kt + 1 < KT) {
      issue_panel(kt + 1, (kt + 1) & 1);       // prefetch next panel
      asm volatile("s_wait_asynccnt 0x4" ::: "memory");   // this panel landed
    } else {
      asm volatile("s_wait_asynccnt 0x0" ::: "memory");
    }
    __syncthreads();                           // all waves' panel writes visible

    if (kt + 2 < KT)                           // L2 lookahead on A stream
      __builtin_prefetch((const void*)(A + (size_t)(m0 + col) * K + (kt + 2) * 32), 0, 1);

    v16h a[4];
#pragma unroll
    for (int mi = 0; mi < 4; ++mi)
      a[mi] = load_a16(A + (size_t)(m0 + mi * 16 + col) * K + kt * 32 + kb);

    const h16* sbuf = &sB[kt & 1][0];
#pragma unroll
    for (int nj = 0; nj < 4; ++nj) {
      v16h b = *(const v16h*)(sbuf + (wx * 4 + nj) * 512 + lane * 16);
#pragma unroll
      for (int mi = 0; mi < 4; ++mi)
        acc[mi][nj] = wmma_f16(a[mi], b, acc[mi][nj]);
    }
    __syncthreads();                           // consumption done before reuse
  }

#pragma unroll
  for (int mi = 0; mi < 4; ++mi)
#pragma unroll
    for (int nj = 0; nj < 4; ++nj) {
      int nc = n0 + wx * 64 + nj * 16 + col;
      float bv = bias ? bias[nc] : 0.0f;
#pragma unroll
      for (int r = 0; r < 8; ++r) {
        int row = m0 + mi * 16 + 8 * g + r;
        Cout[(size_t)row * Ntot + nc] = acc[mi][nj][r] + bv;
      }
    }
}

// ---------------------------------------------------------------------------
// Per-head LayerNorm on q,k + pack:
//   qh/kh : [BH][N][Dh] f16 (row major)  -- LN over Dh applied
//   vt    : [BH][Dh][N] f16 (transposed) -- plain convert
// One wave per (b,n,h). Block = 256 (8 waves).
// ---------------------------------------------------------------------------
__global__ void __launch_bounds__(256) ln_pack_kernel(const float* __restrict__ qkv,
                                                      const float* __restrict__ qg,
                                                      const float* __restrict__ qb,
                                                      const float* __restrict__ kg,
                                                      const float* __restrict__ kbeta,
                                                      h16* __restrict__ qh,
                                                      h16* __restrict__ kh,
                                                      h16* __restrict__ vt) {
  int w    = blockIdx.x * 8 + (threadIdx.x >> 5);
  int lane = threadIdx.x & 31;
  int bn = w >> 4;               // 0..4095
  int h  = w & 15;
  int n  = bn & (NN - 1);
  int bh = (bn >> 11) * HH + h;  // b*16+h
  const float* base = qkv + (size_t)bn * N3C;

#pragma unroll
  for (int which = 0; which < 2; ++which) {   // 0 = q, 1 = k
    const float* p = base + which * CC + h * DH;
    float a0 = p[lane], a1 = p[lane + 32];
    float s  = a0 + a1;
    float ss = a0 * a0 + a1 * a1;
#pragma unroll
    for (int off = 16; off > 0; off >>= 1) {
      s  += __shfl_xor(s,  off, 32);
      ss += __shfl_xor(ss, off, 32);
    }
    float mean = s * (1.0f / 64.0f);
    float var  = ss * (1.0f / 64.0f) - mean * mean;
    float inv  = rsqrtf(var + 1e-6f);
    const float* gam = which ? kg : qg;
    const float* bet = which ? kbeta : qb;
    h16* dst = (which ? kh : qh) + ((size_t)bh * NN + n) * DH;
    dst[lane]      = (h16)(gam[lane]      * (a0 - mean) * inv + bet[lane]);
    dst[lane + 32] = (h16)(gam[lane + 32] * (a1 - mean) * inv + bet[lane + 32]);
  }
  {
    const float* p = base + 2 * CC + h * DH;
    h16* dst = vt + (size_t)bh * DH * NN;
    dst[(size_t)lane * NN + n]        = (h16)p[lane];
    dst[(size_t)(lane + 32) * NN + n] = (h16)p[lane + 32];
  }
}

// ---------------------------------------------------------------------------
// Flash attention, computed transposed so softmax stats are per-lane:
//   S^T = K * Q^T   (rows = keys, cols = queries)
//   O^T = V^T * P^T (rows = d,    cols = queries)
// Each wave handles 32 queries for one (b,h); block = 256 (8 waves).
// ---------------------------------------------------------------------------
__global__ void __launch_bounds__(256) flash_attn_kernel(const h16* __restrict__ qh,
                                                         const h16* __restrict__ kh,
                                                         const h16* __restrict__ vt,
                                                         h16* __restrict__ aoh) {
  const int lane = threadIdx.x & 31;
  const int wave = threadIdx.x >> 5;
  const int bh = blockIdx.x;                   // 0..31 (b*16+h)
  const int q0 = blockIdx.y * 256 + wave * 32; // first query of this wave
  const int col = lane & 15;
  const int g   = lane >> 4;
  const int kbo = g * 16;

  const h16* qbase = qh + (size_t)bh * NN * DH;
  const h16* kbase = kh + (size_t)bh * NN * DH;
  const h16* vbase = vt + (size_t)bh * DH * NN;

  // B-fragments of Q^T: lane col = query, K(=d) in [16g,16g+16) per kk half
  v16h qB[2][2];
#pragma unroll
  for (int nj = 0; nj < 2; ++nj)
#pragma unroll
    for (int kk = 0; kk < 2; ++kk)
      qB[nj][kk] = *(const v16h*)(qbase + (size_t)(q0 + nj * 16 + col) * DH + kk * 32 + g * 16);

  v8f ot[4][2];
#pragma unroll
  for (int di = 0; di < 4; ++di)
#pragma unroll
    for (int nj = 0; nj < 2; ++nj) ot[di][nj] = zero_v8f();

  float mrun[2] = {-1e30f, -1e30f};
  float lrun[2] = {0.0f, 0.0f};

  for (int kt = 0; kt < NN / 32; ++kt) {
    // ---- S^T = K * Q^T over this 32-key tile
    v8f st[2][2];
#pragma unroll
    for (int ki = 0; ki < 2; ++ki)
#pragma unroll
      for (int nj = 0; nj < 2; ++nj) st[ki][nj] = zero_v8f();

#pragma unroll
    for (int kk = 0; kk < 2; ++kk) {
      v16h aK0 = load_a16(kbase + (size_t)(kt * 32 + col) * DH + kk * 32 + kbo);
      v16h aK1 = load_a16(kbase + (size_t)(kt * 32 + 16 + col) * DH + kk * 32 + kbo);
#pragma unroll
      for (int nj = 0; nj < 2; ++nj) {
        st[0][nj] = wmma_f16(aK0, qB[nj][kk], st[0][nj]);
        st[1][nj] = wmma_f16(aK1, qB[nj][kk], st[1][nj]);
      }
    }

    // ---- online softmax per query column; build P^T B-fragments
    v16h pB[2];
#pragma unroll
    for (int nj = 0; nj < 2; ++nj) {
      float tmax = -1e30f;
#pragma unroll
      for (int r = 0; r < 8; ++r)
        tmax = fmaxf(tmax, fmaxf(st[0][nj][r], st[1][nj][r]));
      tmax = fmaxf(tmax, __shfl_xor(tmax, 16, 32));
      tmax *= 0.125f;                         // scale = Dh^-0.5
      float mnew  = fmaxf(mrun[nj], tmax);
      float alpha = __expf(mrun[nj] - mnew);
      mrun[nj] = mnew;

      float p0[8], p1[8];
      float rsum = 0.0f;
#pragma unroll
      for (int r = 0; r < 8; ++r) {
        p0[r] = __expf(st[0][nj][r] * 0.125f - mnew);
        p1[r] = __expf(st[1][nj][r] * 0.125f - mnew);
        rsum += p0[r] + p1[r];
      }
      rsum += __shfl_xor(rsum, 16, 32);
      lrun[nj] = lrun[nj] * alpha + rsum;

      // Exchange with partner lane-group to fill the B-fragment K pattern
      float x0[8], x1[8];
#pragma unroll
      for (int r = 0; r < 8; ++r) {
        x0[r] = __shfl_xor(p0[r], 16, 32);
        x1[r] = __shfl_xor(p1[r], 16, 32);
      }
      v16h pb;
#pragma unroll
      for (int e = 0; e < 8; ++e) {
        float lo = (g == 0) ? p0[e] : x1[e];   // K = 16g + e
        float hi = (g == 0) ? x0[e] : p1[e];   // K = 16g + 8 + e
        pb[e]     = (h16)lo;
        pb[e + 8] = (h16)hi;
      }
      pB[nj] = pb;

      // rescale accumulated O^T columns
#pragma unroll
      for (int di = 0; di < 4; ++di)
#pragma unroll
        for (int r = 0; r < 8; ++r) ot[di][nj][r] *= alpha;
    }

    // ---- O^T += V^T * P^T
#pragma unroll
    for (int di = 0; di < 4; ++di) {
      v16h aV = load_a16(vbase + (size_t)(di * 16 + col) * NN + kt * 32 + kbo);
#pragma unroll
      for (int nj = 0; nj < 2; ++nj)
        ot[di][nj] = wmma_f16(aV, pB[nj], ot[di][nj]);
    }
  }

  // ---- epilogue: normalize and store to [B,N,C] f16 (c = h*64 + d)
  const int b = bh >> 4, h = bh & 15;
#pragma unroll
  for (int nj = 0; nj < 2; ++nj) {
    float inv = 1.0f / lrun[nj];
    int n = q0 + nj * 16 + col;
    size_t rowoff = ((size_t)(b * NN + n)) * CC + h * DH;
#pragma unroll
    for (int di = 0; di < 4; ++di) {
      v8h pk;
#pragma unroll
      for (int r = 0; r < 8; ++r) pk[r] = (h16)(ot[di][nj][r] * inv);
      *(v8h*)(aoh + rowoff + di * 16 + g * 8) = pk;   // d = di*16 + 8g + r
    }
  }
}

// ---------------------------------------------------------------------------
// Host-side launch
// ---------------------------------------------------------------------------
extern "C" void kernel_launch(void* const* d_in, const int* in_sizes, int n_in,
                              void* d_out, int out_size, void* d_ws, size_t ws_size,
                              hipStream_t stream) {
  (void)in_sizes; (void)n_in; (void)out_size; (void)ws_size;
  const float* x      = (const float*)d_in[0];
  const float* qkv_w  = (const float*)d_in[1];
  const float* q_gam  = (const float*)d_in[2];
  const float* q_bet  = (const float*)d_in[3];
  const float* k_gam  = (const float*)d_in[4];
  const float* k_bet  = (const float*)d_in[5];
  const float* proj_w = (const float*)d_in[6];
  const float* proj_b = (const float*)d_in[7];
  float* out = (float*)d_out;

  // Workspace layout (all offsets 256B aligned)
  char* ws = (char*)d_ws;
  size_t off = 0;
  float* qkvf   = (float*)(ws + off); off += (size_t)MROWS * N3C * sizeof(float);   // 50.3 MB
  h16*   xh     = (h16*)(ws + off);   off += (size_t)MROWS * CC * sizeof(h16);      //  8.4 MB
  h16*   wqkvp  = (h16*)(ws + off);   off += (size_t)CC * N3C * sizeof(h16);        //  6.3 MB
  h16*   wprojp = (h16*)(ws + off);   off += (size_t)CC * CC * sizeof(h16);         //  2.1 MB
  h16*   qh     = (h16*)(ws + off);   off += (size_t)BB * HH * NN * DH * sizeof(h16);
  h16*   kh     = (h16*)(ws + off);   off += (size_t)BB * HH * NN * DH * sizeof(h16);
  h16*   vt     = (h16*)(ws + off);   off += (size_t)BB * HH * DH * NN * sizeof(h16);
  h16*   aoh    = (h16*)(ws + off);   off += (size_t)MROWS * CC * sizeof(h16);

  // 1) x -> f16
  {
    int n = MROWS * CC;
    cvt_f32_f16_kernel<<<(n + 255) / 256, 256, 0, stream>>>(x, xh, n);
  }
  // 2) pack weights into B-fragment tiles
  {
    int tiles = (CC / 32) * (N3C / 16);           // 6144
    pack_w_kernel<<<(tiles * 32 + 255) / 256, 256, 0, stream>>>(qkv_w, wqkvp, CC, N3C);
  }
  {
    int tiles = (CC / 32) * (CC / 16);            // 2048
    pack_w_kernel<<<(tiles * 32 + 255) / 256, 256, 0, stream>>>(proj_w, wprojp, CC, CC);
  }
  // 3) QKV GEMM: [4096 x 1024] x [1024 x 3072] -> f32
  {
    dim3 grid(N3C / 128, MROWS / 128);            // 24 x 32
    gemm_wmma_kernel<<<grid, 128, 0, stream>>>(xh, wqkvp, nullptr, qkvf, MROWS, N3C, CC);
  }
  // 4) LayerNorm(q,k) + pack q/k/v for attention
  {
    int waves = MROWS * HH;                       // 65536
    ln_pack_kernel<<<waves / 8, 256, 0, stream>>>(qkvf, q_gam, q_bet, k_gam, k_bet, qh, kh, vt);
  }
  // 5) Flash attention (all-WMMA, online softmax)
  {
    dim3 grid(BB * HH, NN / 256);                 // 32 x 8
    flash_attn_kernel<<<grid, 256, 0, stream>>>(qh, kh, vt, aoh);
  }
  // 6) Projection GEMM + bias -> f32 output
  {
    dim3 grid(CC / 128, MROWS / 128);             // 8 x 32
    gemm_wmma_kernel<<<grid, 128, 0, stream>>>(aoh, wprojp, proj_b, out, MROWS, CC, CC);
  }
}
